// AggregateSet_16535624090064
// MI455X (gfx1250) — compile-verified, matrix operand order checked by reading, not run on hardware
//
#include <hip/hip_runtime.h>

typedef __attribute__((ext_vector_type(16))) __bf16 v16bf;
typedef __attribute__((ext_vector_type(8)))  float  v8f;

union Frag {
    uint4  q[2];
    v16bf  v;
    __bf16 h[16];
};

__device__ __forceinline__ __bf16 tobf(float f) { return (__bf16)f; }  // native RTNE convert

// ---------------------------------------------------------------------------
// Prep: convert fp32 weights to bf16, transposed so each WMMA B-fragment is
// two contiguous 16-byte loads per lane (row = output column, cols = K).
//   WsubT : [256][64]   WqT/WkT/WvT : [512][256]
// ---------------------------------------------------------------------------
__global__ void prep_weights(const float* __restrict__ Wsub,
                             const float* __restrict__ Wq,
                             const float* __restrict__ Wk,
                             const float* __restrict__ Wv,
                             __bf16* __restrict__ WsubT,
                             __bf16* __restrict__ WqT,
                             __bf16* __restrict__ WkT,
                             __bf16* __restrict__ WvT)
{
    int idx = blockIdx.x * blockDim.x + threadIdx.x;
    if (idx < 64 * 256) {
        int n = idx >> 6, k = idx & 63;              // Wsub is (64,256) row-major
        WsubT[idx] = tobf(Wsub[k * 256 + n]);
    }
    if (idx < 256 * 512) {
        int j = idx >> 8, k = idx & 255;             // Wq/k/v are (256,512) row-major
        WqT[idx] = tobf(Wq[k * 512 + j]);
        WkT[idx] = tobf(Wk[k * 512 + j]);
        WvT[idx] = tobf(Wv[k * 512 + j]);
    }
}

// ---------------------------------------------------------------------------
// Main: one workgroup per batch element. 256 threads = 8 waves; wave w owns
// M-tile w (16 rows of the 128-element set).
// ---------------------------------------------------------------------------
#define SA 264   // LDS activ row stride (bf16 elems); 264*2B = 528B -> bank-skewed

__global__ __launch_bounds__(256)
void aggregate_kernel(const float* __restrict__ x,
                      const float* __restrict__ b_sub,
                      const float* __restrict__ b_q,
                      const float* __restrict__ b_k,
                      const float* __restrict__ b_v,
                      const __bf16* __restrict__ WsubT,
                      const __bf16* __restrict__ WqT,
                      const __bf16* __restrict__ WkT,
                      const __bf16* __restrict__ WvT,
                      float* __restrict__ out)
{
    __shared__ __align__(16) __bf16 sActiv[128 * SA];  // 66 KB bf16 activations
    __shared__ float sScore[128];
    __shared__ float sW[128];
    __shared__ float sMask[128];
    __shared__ float sRed[128];
    __shared__ float sPooled[64];

    const int b     = blockIdx.x;
    const int t     = threadIdx.x;
    const int lane  = t & 31;
    const int wave  = t >> 5;        // == M-tile index (0..7)
    const int nsel  = lane & 15;     // column / row-in-tile selector
    const int khalf = lane >> 4;     // which K-half this lane holds
    const int mloc  = khalf * 8;     // C/D row base for this half

    const long xbase = (long)b * 8320;   // 128*64 feats + 128 mask

    // ---- mask + elem_frac ----
    if (t < 128) sMask[t] = x[xbase + 8192 + t];
    __syncthreads();
    if (t < 128) sRed[t] = sMask[t];
    __syncthreads();
    for (int s = 64; s > 0; s >>= 1) { if (t < s) sRed[t] += sRed[t + s]; __syncthreads(); }
    if (t == 0) out[(long)b * 513 + 512] = sRed[0] * (1.0f / 128.0f);

    // ---- GEMM1: activ = xe @ W_sub + b_sub  ->  sActiv (bf16) ----
    // A fragments straight from global x (fp32 -> bf16), hoisted for 16 N-tiles.
    Frag a1[2];
    {
        const int m = wave * 16 + nsel;
        const float* xr = x + xbase + (long)m * 64;
        #pragma unroll
        for (int kt = 0; kt < 2; ++kt) {
            const int k0 = kt * 32 + khalf * 8;
            #pragma unroll
            for (int i = 0; i < 8; ++i) a1[kt].h[i]     = tobf(xr[k0 + i]);
            #pragma unroll
            for (int i = 0; i < 8; ++i) a1[kt].h[8 + i] = tobf(xr[k0 + 16 + i]);
        }
    }
    for (int nt = 0; nt < 16; ++nt) {
        v8f acc = {};
        const int j = nt * 16 + nsel;               // row of WsubT
        #pragma unroll
        for (int kt = 0; kt < 2; ++kt) {
            const int k0 = kt * 32 + khalf * 8;
            Frag bf;
            bf.q[0] = *(const uint4*)&WsubT[j * 64 + k0];
            bf.q[1] = *(const uint4*)&WsubT[j * 64 + k0 + 16];
            acc = __builtin_amdgcn_wmma_f32_16x16x32_bf16(false, a1[kt].v, false, bf.v,
                                                          (short)0, acc, false, false);
        }
        const float bias = b_sub[j];
        #pragma unroll
        for (int r = 0; r < 8; ++r) {
            const int m = wave * 16 + mloc + r;
            sActiv[m * SA + j] = tobf(acc[r] + bias);
        }
    }
    __syncthreads();

    // ---- hoist activ A-fragments (reused by q,k,v GEMMs across all 8 heads) ----
    Frag afr[8];
    {
        const int m = wave * 16 + nsel;
        #pragma unroll
        for (int kt = 0; kt < 8; ++kt) {
            const int k0 = kt * 32 + khalf * 8;
            afr[kt].q[0] = *(const uint4*)&sActiv[m * SA + k0];
            afr[kt].q[1] = *(const uint4*)&sActiv[m * SA + k0 + 16];
        }
    }

    for (int h = 0; h < 8; ++h) {
        // ---- score pass: q,k tiles share the A fragment; dot formed in-register ----
        float prodacc[8] = {0.f, 0.f, 0.f, 0.f, 0.f, 0.f, 0.f, 0.f};
        for (int nt = 0; nt < 4; ++nt) {
            v8f qa = {}, ka = {};
            const int j = h * 64 + nt * 16 + nsel;  // row of WqT/WkT
            #pragma unroll
            for (int kt = 0; kt < 8; ++kt) {
                const int k0 = kt * 32 + khalf * 8;
                Frag bq, bk;
                bq.q[0] = *(const uint4*)&WqT[j * 256 + k0];
                bq.q[1] = *(const uint4*)&WqT[j * 256 + k0 + 16];
                bk.q[0] = *(const uint4*)&WkT[j * 256 + k0];
                bk.q[1] = *(const uint4*)&WkT[j * 256 + k0 + 16];
                qa = __builtin_amdgcn_wmma_f32_16x16x32_bf16(false, afr[kt].v, false, bq.v,
                                                             (short)0, qa, false, false);
                ka = __builtin_amdgcn_wmma_f32_16x16x32_bf16(false, afr[kt].v, false, bk.v,
                                                             (short)0, ka, false, false);
            }
            const float bqv = b_q[j];
            const float bkv = b_k[j];
            #pragma unroll
            for (int r = 0; r < 8; ++r)
                prodacc[r] += (qa[r] + bqv) * (ka[r] + bkv);
        }
        // reduce over the 16 n-lanes of each half (rows stay separated by khalf)
        #pragma unroll
        for (int r = 0; r < 8; ++r) {
            float s = prodacc[r];
            s += __shfl_xor(s, 1, 32);
            s += __shfl_xor(s, 2, 32);
            s += __shfl_xor(s, 4, 32);
            s += __shfl_xor(s, 8, 32);
            prodacc[r] = s;
        }
        if (nsel == 0) {
            #pragma unroll
            for (int r = 0; r < 8; ++r)
                sScore[wave * 16 + mloc + r] = prodacc[r];
        }
        __syncthreads();

        // ---- masked softmax over the 128 set elements (block-wide) ----
        float z = 0.f, mk = 0.f;
        if (t < 128) {
            mk = sMask[t];
            z  = (mk > 0.f) ? sScore[t] * 0.125f : 0.f;   // /sqrt(64)
            sRed[t] = fmaxf(z, 0.f);                      // relu(z*mask)
        }
        __syncthreads();
        for (int s = 64; s > 0; s >>= 1) { if (t < s) sRed[t] = fmaxf(sRed[t], sRed[t + s]); __syncthreads(); }
        const float zmax = sRed[0];
        __syncthreads();
        float me = 0.f;
        if (t < 128) { me = mk * __expf(z - zmax); sRed[t] = me; }
        __syncthreads();
        for (int s = 64; s > 0; s >>= 1) { if (t < s) sRed[t] += sRed[t + s]; __syncthreads(); }
        const float denom = sRed[0] + 1.0f;
        if (t < 128) sW[t] = me / denom;
        if (t < 64)  sPooled[t] = 0.f;
        __syncthreads();

        // ---- v pass + attention pooling (LDS float atomics) ----
        for (int nt = 0; nt < 4; ++nt) {
            v8f va = {};
            const int j = h * 64 + nt * 16 + nsel;
            #pragma unroll
            for (int kt = 0; kt < 8; ++kt) {
                const int k0 = kt * 32 + khalf * 8;
                Frag bv;
                bv.q[0] = *(const uint4*)&WvT[j * 256 + k0];
                bv.q[1] = *(const uint4*)&WvT[j * 256 + k0 + 16];
                va = __builtin_amdgcn_wmma_f32_16x16x32_bf16(false, afr[kt].v, false, bv.v,
                                                             (short)0, va, false, false);
            }
            const float bvv = b_v[j];
            float p = 0.f;
            #pragma unroll
            for (int r = 0; r < 8; ++r) {
                const int m = wave * 16 + mloc + r;
                p += sW[m] * (va[r] + bvv);
            }
            p += __shfl_xor(p, 16, 32);          // join the two khalf row-groups
            if (lane < 16) atomicAdd(&sPooled[nt * 16 + lane], p);
        }
        __syncthreads();
        if (t < 64) out[(long)b * 513 + h * 64 + t] = sPooled[t];
        __syncthreads();
    }
}

// ---------------------------------------------------------------------------
extern "C" void kernel_launch(void* const* d_in, const int* in_sizes, int n_in,
                              void* d_out, int out_size, void* d_ws, size_t ws_size,
                              hipStream_t stream)
{
    const float* x     = (const float*)d_in[0];
    const float* Wsub  = (const float*)d_in[1];
    const float* bsub  = (const float*)d_in[2];
    const float* Wq    = (const float*)d_in[3];
    const float* bq    = (const float*)d_in[4];
    const float* Wk    = (const float*)d_in[5];
    const float* bk    = (const float*)d_in[6];
    const float* Wv    = (const float*)d_in[7];
    const float* bv    = (const float*)d_in[8];
    float* out = (float*)d_out;

    const int B = in_sizes[0] / 8320;   // 128*64 + 128 per set

    __bf16* WsubT = (__bf16*)d_ws;          // 256*64
    __bf16* WqT   = WsubT + 256 * 64;       // 512*256
    __bf16* WkT   = WqT   + 512 * 256;
    __bf16* WvT   = WkT   + 512 * 256;

    prep_weights<<<(256 * 512 + 255) / 256, 256, 0, stream>>>(
        Wsub, Wq, Wk, Wv, WsubT, WqT, WkT, WvT);

    aggregate_kernel<<<B, 256, 0, stream>>>(
        x, bsub, bq, bk, bv, WsubT, WqT, WkT, WvT, out);
}
